// MultiHeadedAttention_17626545783264
// MI455X (gfx1250) — matrix operand order repaired and checked by reading, not compiled, not run on hardware
//
#include <hip/hip_runtime.h>
#include <cstdint>
#include <cstddef>

// ---------- constants for this problem ----------
namespace {
constexpr int kB = 4, kS = 2048, kD = 1024, kH = 16, kDK = 64;
}

using u16 = unsigned short;
typedef u16    u16x8   __attribute__((ext_vector_type(8)));
typedef __bf16 bf16x16 __attribute__((ext_vector_type(16)));
typedef float  f32x8   __attribute__((ext_vector_type(8)));
typedef int    v4i     __attribute__((vector_size(16)));   // matches builtin proto

#if __has_builtin(__builtin_amdgcn_global_load_async_to_lds_b128) && \
    __has_builtin(__builtin_amdgcn_s_wait_asynccnt)
#define HAS_ASYNC_LDS 1
#else
#define HAS_ASYNC_LDS 0
#endif

__device__ __forceinline__ u16 f2bf(float f) {
  uint32_t u = __builtin_bit_cast(uint32_t, f);
  u += 0x7FFFu + ((u >> 16) & 1u);   // round-to-nearest-even
  return (u16)(u >> 16);
}
__device__ __forceinline__ __bf16 bfbits(u16 v) {
  return __builtin_bit_cast(__bf16, v);
}
__device__ __forceinline__ f32x8 wmma_bf16(bf16x16 a, bf16x16 b, f32x8 c) {
  // (neg_a, A, neg_b, B, c_mod, C, reuse_a, reuse_b)
  return __builtin_amdgcn_wmma_f32_16x16x32_bf16(false, a, false, b, (short)0, c,
                                                 false, false);
}

// 16-byte global -> LDS copy. Async (ASYNCcnt-tracked, bypasses VGPRs) when the
// gfx1250 builtin exists; otherwise synchronous load+ds_store fallback.
__device__ __forceinline__ void cp16(u16* lds_dst, const u16* gsrc) {
#if HAS_ASYNC_LDS
  __builtin_amdgcn_global_load_async_to_lds_b128(
      (__attribute__((address_space(1))) v4i*)gsrc,
      (__attribute__((address_space(3))) v4i*)lds_dst, 0, 0);
#else
  *(u16x8*)lds_dst = *(const u16x8*)gsrc;
#endif
}

// ---------- fp32 -> bf16 conversion ----------
__global__ void cvt_bf16_kernel(const float* __restrict__ src,
                                u16* __restrict__ dst, int n) {
  for (int i = blockIdx.x * blockDim.x + threadIdx.x; i < n;
       i += gridDim.x * blockDim.x)
    dst[i] = f2bf(src[i]);
}

// ---------- tiled GEMM: out[m,n] = sum_k A[m,k] * W[n,k] + bias[n] ----------
// A: [M,K] bf16, W: [N,K] bf16 (both K-contiguous). 128x128 block tile, K-step
// 32, double-buffered LDS staged with async-to-LDS copies. 8 waves; wave
// (wr,wc) in 2x4 grid owns a 64x32 patch = 4x2 tiles of 16x16.
__global__ __launch_bounds__(256) void gemm_bf16_kernel(
    const u16* __restrict__ A, const u16* __restrict__ W,
    const float* __restrict__ bias, u16* __restrict__ outb,
    float* __restrict__ outf, int M, int N, int K) {
  __shared__ u16 lA[2][128][40];  // pad 8 -> 80B row stride (16B aligned)
  __shared__ u16 lB[2][128][40];

  const int tid  = threadIdx.x;
  const int lane = tid & 31;
  const int wave = tid >> 5;
  const int m0 = blockIdx.x * 128;
  const int n0 = blockIdx.y * 128;
  const int wr = wave >> 2;          // 0..1
  const int wc = wave & 3;           // 0..3

  // ISA wave32 16-bit operand lane decomposition
  const int am  = lane & 15;         // A: row M within tile
  const int ak  = (lane >> 4) << 3;  // A: K half offset (0/8)
  const int bn  = lane & 15;         // B: col N within tile
  const int bk  = (lane >> 4) << 3;  // B: K half offset (0/8)
  const int crg = (lane >> 4) << 3;  // C: row group (0/8)
  const int cn  = lane & 15;         // C: col

  const f32x8 zero = {0.f, 0.f, 0.f, 0.f, 0.f, 0.f, 0.f, 0.f};
  f32x8 acc[4][2];
#pragma unroll
  for (int i = 0; i < 4; ++i)
#pragma unroll
    for (int j = 0; j < 2; ++j) acc[i][j] = zero;

  const int lr = tid >> 1;           // 0..127 (row loaded by this thread)
  const int lc = (tid & 1) * 16;     // 0 / 16

  // issue the 4 copies for one 128x32 A tile + 128x32 W tile
  auto stage_tile = [&](int buf, int kt) {
    cp16(&lA[buf][lr][lc],     A + (size_t)(m0 + lr) * K + kt + lc);
    cp16(&lA[buf][lr][lc + 8], A + (size_t)(m0 + lr) * K + kt + lc + 8);
    cp16(&lB[buf][lr][lc],     W + (size_t)(n0 + lr) * K + kt + lc);
    cp16(&lB[buf][lr][lc + 8], W + (size_t)(n0 + lr) * K + kt + lc + 8);
  };

  const int nk = K >> 5;
  stage_tile(0, 0);  // prologue

  for (int t = 0; t < nk; ++t) {
    const int buf = t & 1;
    if (t + 1 < nk) {
      stage_tile(buf ^ 1, (t + 1) << 5);  // prefetch next tile (async)
#if HAS_ASYNC_LDS
      __builtin_amdgcn_s_wait_asynccnt(4);  // retire current tile's 4 copies
#endif
    } else {
#if HAS_ASYNC_LDS
      __builtin_amdgcn_s_wait_asynccnt(0);  // epilogue: drain everything
#endif
    }
    __syncthreads();

    bf16x16 bfr[2];
#pragma unroll
    for (int j = 0; j < 2; ++j) {
      const int n = wc * 32 + j * 16 + bn;
#pragma unroll
      for (int e = 0; e < 16; ++e) bfr[j][e] = bfbits(lB[buf][n][bk + e + ((e >= 8) ? 8 : 0)]);
    }
#pragma unroll
    for (int i = 0; i < 4; ++i) {
      const int m = wr * 64 + i * 16 + am;
      bf16x16 af;
#pragma unroll
      for (int vi = 0; vi < 8; ++vi) {
        const int ka = ((vi >= 4) ? 16 : 0) + ak + 2 * (vi & 3);
        af[2 * vi]     = bfbits(lA[buf][m][ka]);
        af[2 * vi + 1] = bfbits(lA[buf][m][ka + 1]);
      }
#pragma unroll
      for (int j = 0; j < 2; ++j) acc[i][j] = wmma_bf16(af, bfr[j], acc[i][j]);
    }
    __syncthreads();
  }

#pragma unroll
  for (int i = 0; i < 4; ++i) {
#pragma unroll
    for (int j = 0; j < 2; ++j) {
      const int n = n0 + wc * 32 + j * 16 + cn;
      const float bvv = bias[n];
#pragma unroll
      for (int r = 0; r < 8; ++r) {
        const int m = m0 + wr * 64 + i * 16 + crg + r;
        const float v = acc[i][j][r] + bvv;
        if (outf) outf[(size_t)m * N + n] = v;
        else      outb[(size_t)m * N + n] = f2bf(v);
      }
    }
  }
}

// ---------- flash-style attention ----------
// One block per (b, h, 64-query tile). Loops over 32 key blocks of 64.
// Reference semantics: masked scores = 1e-13 (NOT -inf), then softmax.
__global__ __launch_bounds__(256) void attn_kernel(
    const u16* __restrict__ Q, const u16* __restrict__ K,
    const u16* __restrict__ V, const int* __restrict__ mask,
    u16* __restrict__ ctx) {
  __shared__ u16   lQ[64][72];
  __shared__ u16   lK[64][72];
  __shared__ u16   lV[64][72];
  __shared__ float lS[64][68];
  __shared__ u16   lP[64][72];
  __shared__ float rowM[64], rowL[64], rowF[64];

  const int tid  = threadIdx.x;
  const int lane = tid & 31;
  const int wave = tid >> 5;
  const int gid = blockIdx.x;
  const int qt = gid & 31;            // 32 q-tiles
  const int h  = (gid >> 5) & 15;
  const int b  = gid >> 9;

  const size_t headoff = (size_t)h * kDK;
  const int q0 = qt * 64;

  const int am  = lane & 15;
  const int ak  = (lane >> 4) << 3;
  const int bn  = lane & 15;
  const int bk  = (lane >> 4) << 3;
  const int crg = (lane >> 4) << 3;
  const int cn  = lane & 15;

  const int r  = tid >> 2;            // staging row (0..63)
  const int c0 = (tid & 3) * 16;      // staging column (0/16/32/48)

  // load Q tile (64 rows x 64 head-dims), async
  cp16(&lQ[r][c0],     Q + (size_t)(b * kS + q0 + r) * kD + headoff + c0);
  cp16(&lQ[r][c0 + 8], Q + (size_t)(b * kS + q0 + r) * kD + headoff + c0 + 8);
  if (tid < 64) { rowM[tid] = -3.0e38f; rowL[tid] = 0.f; }

  const int t0 = wave * 2;   // this wave's 2 tiles in the 4x4 tile grid
  const f32x8 zero = {0.f, 0.f, 0.f, 0.f, 0.f, 0.f, 0.f, 0.f};
  f32x8 oacc[2];
  oacc[0] = zero; oacc[1] = zero;

  for (int kb = 0; kb < 32; ++kb) {
    const int k0 = kb * 64;
    {  // load K/V key block, async
      const u16* gk = K + (size_t)(b * kS + k0 + r) * kD + headoff + c0;
      const u16* gv = V + (size_t)(b * kS + k0 + r) * kD + headoff + c0;
      cp16(&lK[r][c0], gk);
      cp16(&lK[r][c0 + 8], gk + 8);
      cp16(&lV[r][c0], gv);
      cp16(&lV[r][c0 + 8], gv + 8);
      if (kb + 1 < 32) {  // global_prefetch_b8 for next key block
        __builtin_prefetch(K + (size_t)(b * kS + k0 + 64 + r) * kD + headoff + c0, 0, 1);
        __builtin_prefetch(V + (size_t)(b * kS + k0 + 64 + r) * kD + headoff + c0, 0, 1);
      }
    }
#if HAS_ASYNC_LDS
    __builtin_amdgcn_s_wait_asynccnt(0);
#endif
    __syncthreads();

    // scores = Q * K^T (per wave: 2 tiles of 16x16 out of the 4x4 grid)
#pragma unroll
    for (int t = 0; t < 2; ++t) {
      const int tm = (t0 + t) >> 2, tn = (t0 + t) & 3;
      f32x8 sc = zero;
#pragma unroll
      for (int kk = 0; kk < 64; kk += 32) {
        bf16x16 af, bf;
#pragma unroll
        for (int vi = 0; vi < 8; ++vi) {
          const int ka = kk + ((vi >= 4) ? 16 : 0) + ak + 2 * (vi & 3);
          af[2 * vi]     = bfbits(lQ[tm * 16 + am][ka]);
          af[2 * vi + 1] = bfbits(lQ[tm * 16 + am][ka + 1]);
        }
#pragma unroll
        for (int e = 0; e < 16; ++e)
          bf[e] = bfbits(lK[tn * 16 + bn][kk + bk + e + ((e >= 8) ? 8 : 0)]);   // B[d][n] = K[n][d]
        sc = wmma_bf16(af, bf, sc);
      }
      // scale + non-standard mask fill (TINY, not -inf), spill to LDS
#pragma unroll
      for (int rr = 0; rr < 8; ++rr) {
        const int qr = tm * 16 + crg + rr;
        const int kc = tn * 16 + cn;
        const int mv = mask[((size_t)b * kS + (q0 + qr)) * kS + (k0 + kc)];
        lS[qr][kc] = (mv == 0) ? 1e-13f : sc[rr] * 0.125f;  // 1/sqrt(64)
      }
    }
    __syncthreads();

    // online softmax row pass (one thread per query row)
    if (tid < 64) {
      const int rw = tid;
      const float mold = rowM[rw];
      float bmax = lS[rw][0];
      for (int c = 1; c < 64; ++c) bmax = fmaxf(bmax, lS[rw][c]);
      const float mnew = fmaxf(mold, bmax);
      const float f = __expf(mold - mnew);
      float s = 0.f;
      for (int c = 0; c < 64; ++c) {
        const float p = __expf(lS[rw][c] - mnew);
        lP[rw][c] = f2bf(p);
        s += p;
      }
      rowM[rw] = mnew;
      rowL[rw] = rowL[rw] * f + s;
      rowF[rw] = f;
    }
    __syncthreads();

    // O = O * rowF + P * V
#pragma unroll
    for (int t = 0; t < 2; ++t) {
      const int tm = (t0 + t) >> 2, tn = (t0 + t) & 3;
#pragma unroll
      for (int rr = 0; rr < 8; ++rr)
        oacc[t][rr] = oacc[t][rr] * rowF[tm * 16 + crg + rr];
#pragma unroll
      for (int kk = 0; kk < 64; kk += 32) {
        bf16x16 af, bf;
#pragma unroll
        for (int vi = 0; vi < 8; ++vi) {
          const int ka = kk + ((vi >= 4) ? 16 : 0) + ak + 2 * (vi & 3);
          af[2 * vi]     = bfbits(lP[tm * 16 + am][ka]);
          af[2 * vi + 1] = bfbits(lP[tm * 16 + am][ka + 1]);
        }
#pragma unroll
        for (int e = 0; e < 16; ++e)
          bf[e] = bfbits(lV[kk + bk + e + ((e >= 8) ? 8 : 0)][tn * 16 + bn]);   // B[k][d] = V[k][d]
        oacc[t] = wmma_bf16(af, bf, oacc[t]);
      }
    }
    __syncthreads();
  }

  // finalize: divide by row sums, write bf16 context [B,S,D]
#pragma unroll
  for (int t = 0; t < 2; ++t) {
    const int tm = (t0 + t) >> 2, tn = (t0 + t) & 3;
#pragma unroll
    for (int rr = 0; rr < 8; ++rr) {
      const int qr = tm * 16 + crg + rr;
      const float invl = 1.0f / rowL[qr];
      const int dc = tn * 16 + cn;
      ctx[(size_t)(b * kS + q0 + qr) * kD + h * kDK + dc] =
          f2bf(oacc[t][rr] * invl);
    }
  }
}

// ---------- host launcher ----------
extern "C" void kernel_launch(void* const* d_in, const int* in_sizes, int n_in,
                              void* d_out, int out_size, void* d_ws,
                              size_t ws_size, hipStream_t stream) {
  (void)in_sizes; (void)n_in; (void)out_size; (void)ws_size;
  const float* q   = (const float*)d_in[0];
  const float* k   = (const float*)d_in[1];
  const float* v   = (const float*)d_in[2];
  const int*   msk = (const int*)d_in[3];
  const float* Wq  = (const float*)d_in[4];
  const float* bq  = (const float*)d_in[5];
  const float* Wk  = (const float*)d_in[6];
  const float* bk  = (const float*)d_in[7];
  const float* Wv  = (const float*)d_in[8];
  const float* bv  = (const float*)d_in[9];
  const float* Wo  = (const float*)d_in[10];
  const float* bo  = (const float*)d_in[11];
  float* out = (float*)d_out;

  constexpr size_t BSD = (size_t)kB * kS * kD;   // 8,388,608
  constexpr size_t DDs = (size_t)kD * kD;        // 1,048,576

  char* p = (char*)d_ws;
  u16* xq  = (u16*)p; p += BSD * 2;
  u16* xk  = (u16*)p; p += BSD * 2;
  u16* xv  = (u16*)p; p += BSD * 2;
  u16* wqb = (u16*)p; p += DDs * 2;
  u16* wkb = (u16*)p; p += DDs * 2;
  u16* wvb = (u16*)p; p += DDs * 2;
  u16* wob = (u16*)p; p += DDs * 2;
  u16* qb  = (u16*)p; p += BSD * 2;
  u16* kbb = (u16*)p; p += BSD * 2;
  u16* vbb = (u16*)p; p += BSD * 2;
  u16* ctx = (u16*)p; p += BSD * 2;

  dim3 blk(256);
  cvt_bf16_kernel<<<2048, blk, 0, stream>>>(q, xq, (int)BSD);
  cvt_bf16_kernel<<<2048, blk, 0, stream>>>(k, xk, (int)BSD);
  cvt_bf16_kernel<<<2048, blk, 0, stream>>>(v, xv, (int)BSD);
  cvt_bf16_kernel<<<512,  blk, 0, stream>>>(Wq, wqb, (int)DDs);
  cvt_bf16_kernel<<<512,  blk, 0, stream>>>(Wk, wkb, (int)DDs);
  cvt_bf16_kernel<<<512,  blk, 0, stream>>>(Wv, wvb, (int)DDs);
  cvt_bf16_kernel<<<512,  blk, 0, stream>>>(Wo, wob, (int)DDs);

  dim3 gg(64, 8);  // M/128 x N/128
  gemm_bf16_kernel<<<gg, blk, 0, stream>>>(xq, wqb, bq, qb,  nullptr, kB * kS, kD, kD);
  gemm_bf16_kernel<<<gg, blk, 0, stream>>>(xk, wkb, bk, kbb, nullptr, kB * kS, kD, kD);
  gemm_bf16_kernel<<<gg, blk, 0, stream>>>(xv, wvb, bv, vbb, nullptr, kB * kS, kD, kD);

  attn_kernel<<<kB * kH * (kS / 64), blk, 0, stream>>>(qb, kbb, vbb, msk, ctx);

  gemm_bf16_kernel<<<gg, blk, 0, stream>>>(ctx, wob, bo, nullptr, out, kB * kS, kD, kD);
}
